// Block_50697793962113
// MI455X (gfx1250) — matrix operand order repaired and checked by reading, not compiled, not run on hardware
//
#include <hip/hip_runtime.h>
#include <hip/hip_bf16.h>

// ---------------------------------------------------------------------------
// Transformer block (pre-norm attn + MLP) for MI455X / gfx1250, wave32.
// bf16 WMMA (f32 accum) for all matmuls; flash attention; async global->LDS
// tile copies (ASYNCcnt) with double-buffered LDS.
// ---------------------------------------------------------------------------

typedef __bf16 bf16_t;
typedef __attribute__((ext_vector_type(16))) __bf16 v16bf;
typedef __attribute__((ext_vector_type(8)))  float  v8f;

#define TOK    8192   // B*N tokens
#define SEQ    4096
#define CDIM   512
#define NHEAD  8
#define HDIM   64
#define HIDDEN 2048
#define QKV_LD 1536   // q | k | v packed per token

// ---- async global->LDS (CDNA5 GLOBAL_LOAD_ASYNC_TO_LDS_B128), guarded ------
#if defined(__has_builtin)
#if __has_builtin(__builtin_amdgcn_global_load_async_to_lds_b128) && \
    __has_builtin(__builtin_amdgcn_s_wait_asynccnt)
#define USE_ASYNC_LDS 1
#endif
#endif
#ifndef USE_ASYNC_LDS
#define USE_ASYNC_LDS 0
#endif

// Builtin prototype (from hipcc diagnostic): param1 is
// '__attribute__((__vector_size__(4*sizeof(int)))) int __device__ *' (AS1 int4*).
typedef int v4i __attribute__((vector_size(16)));
typedef __attribute__((address_space(1))) v4i* g_v4i_p;  // global (__device__)
typedef __attribute__((address_space(3))) v4i* l_v4i_p;  // LDS (__shared__)

__device__ __forceinline__ void copy16_to_lds(bf16_t* dst, const bf16_t* src) {
#if USE_ASYNC_LDS
  __builtin_amdgcn_global_load_async_to_lds_b128((g_v4i_p)src, (l_v4i_p)dst, 0, 0);
#else
  *(uint4*)dst = *(const uint4*)src;
#endif
}

__device__ __forceinline__ void async_wait_all() {
#if USE_ASYNC_LDS
  __builtin_amdgcn_s_wait_asynccnt(0);
#endif
}

__device__ __forceinline__ v8f wmma_bf(v16bf a, v16bf b, v8f c) {
  return __builtin_amdgcn_wmma_f32_16x16x32_bf16(false, a, false, b, (short)0, c,
                                                 false, false);
}

// A-matrix 16x32 bf16 fragment (ISA 7.12.2): lane holds row M=lane%16;
// lanes 0-15 -> K {0..7, 16..23}, lanes 16-31 -> K {8..15, 24..31}.
__device__ __forceinline__ v16bf load_frag_a(const bf16_t* base, int stride, int lane) {
  const bf16_t* p = base + (lane & 15) * stride + ((lane & 16) ? 8 : 0);
  union { uint4 u[2]; v16bf v; } t;
  t.u[0] = *(const uint4*)(p);
  t.u[1] = *(const uint4*)(p + 16);
  return t.v;
}

// B-matrix 32x16 bf16 fragment: lane holds col N=lane%16; lanes 0-15 -> K {0..7, 16..23},
// lanes 16-31 -> K {8..15, 24..31}. `base` points at K-contiguous [n][k] storage.
__device__ __forceinline__ v16bf load_frag_b(const bf16_t* base, int stride, int lane) {
  const bf16_t* p = base + (lane & 15) * stride + ((lane & 16) ? 8 : 0);
  union { uint4 u[2]; v16bf v; } t;
  t.u[0] = *(const uint4*)(p);
  t.u[1] = *(const uint4*)(p + 16);
  return t.v;
}

// ---------------------------------------------------------------------------
// Weight convert + transpose: w[K,N] f32 -> wt[N,K] bf16 (once per launch).
// ---------------------------------------------------------------------------
__global__ void cvt_transpose_kernel(const float* __restrict__ w,
                                     bf16_t* __restrict__ wt, int K, int N) {
  int idx = blockIdx.x * blockDim.x + threadIdx.x;
  if (idx >= K * N) return;
  int k = idx % K;  // consecutive threads -> consecutive k (coalesced write)
  int n = idx / K;
  wt[(size_t)n * K + k] = (bf16_t)w[(size_t)k * N + n];
}

// V transpose: qkv v-slice -> vt[b][h][d][n] bf16.
__global__ __launch_bounds__(256) void v_transpose_kernel(
    const bf16_t* __restrict__ qkv, bf16_t* __restrict__ vt) {
  size_t idx = (size_t)blockIdx.x * blockDim.x + threadIdx.x;  // 2*8*64*4096
  int n = (int)(idx & (SEQ - 1));
  int d = (int)((idx >> 12) & (HDIM - 1));
  int h = (int)((idx >> 18) & (NHEAD - 1));
  int b = (int)(idx >> 21);
  vt[idx] = qkv[((size_t)b * SEQ + n) * QKV_LD + 2 * CDIM + h * HDIM + d];
}

// ---------------------------------------------------------------------------
// LayerNorm over CDIM=512, one row per 256-thread block; bf16 output.
// ---------------------------------------------------------------------------
__global__ __launch_bounds__(256) void ln_kernel(const float* __restrict__ x,
                                                 const float* __restrict__ w,
                                                 const float* __restrict__ b,
                                                 bf16_t* __restrict__ out) {
  __shared__ float red[8];
  const int row = blockIdx.x;
  const int t = threadIdx.x;
  const float* xr = x + (size_t)row * CDIM;
  float2 v = *(const float2*)(xr + t * 2);
  float s = v.x + v.y;
#pragma unroll
  for (int m = 16; m; m >>= 1) s += __shfl_xor(s, m, 32);
  if ((t & 31) == 0) red[t >> 5] = s;
  __syncthreads();
  float tot = 0.f;
#pragma unroll
  for (int i = 0; i < 8; ++i) tot += red[i];
  const float mean = tot * (1.0f / CDIM);
  const float dx = v.x - mean, dy = v.y - mean;
  float vs = dx * dx + dy * dy;
#pragma unroll
  for (int m = 16; m; m >>= 1) vs += __shfl_xor(vs, m, 32);
  __syncthreads();
  if ((t & 31) == 0) red[t >> 5] = vs;
  __syncthreads();
  float vtot = 0.f;
#pragma unroll
  for (int i = 0; i < 8; ++i) vtot += red[i];
  const float rstd = rsqrtf(vtot * (1.0f / CDIM) + 1e-5f);
  const float o0 = dx * rstd * w[t * 2 + 0] + b[t * 2 + 0];
  const float o1 = dy * rstd * w[t * 2 + 1] + b[t * 2 + 1];
  out[(size_t)row * CDIM + t * 2 + 0] = (bf16_t)o0;
  out[(size_t)row * CDIM + t * 2 + 1] = (bf16_t)o1;
}

// ---------------------------------------------------------------------------
// bf16 WMMA GEMM: C[M,N](ldc) = act(A[M,K] * Bt[N,K]^T + bias + resid).
// Block tile 128x128, K-step 64, 8 waves (2x4), double-buffered async LDS.
// ---------------------------------------------------------------------------
template <bool OUT_BF16, bool RELU, bool HAS_BIAS, bool HAS_RES>
__global__ __launch_bounds__(256, 2) void gemm_bf16_kernel(
    const bf16_t* __restrict__ A, const bf16_t* __restrict__ Bt,
    void* __restrict__ Cout, const float* __restrict__ bias,
    const float* __restrict__ resid, int M, int N, int K, int ldc) {
  __shared__ __align__(16) bf16_t As[2][128 * 72];  // [row][k0..63], pad 72
  __shared__ __align__(16) bf16_t Bs[2][128 * 72];  // [col][k0..63], pad 72

  const int lane = threadIdx.x & 31;
  const int wave = threadIdx.x >> 5;
  const int wm = wave >> 2;  // 0..1  -> 64 rows
  const int wn = wave & 3;   // 0..3  -> 32 cols
  const int m0 = blockIdx.y * 128;
  const int n0 = blockIdx.x * 128;

  auto load_tiles = [&](int k0, int buf) {
#pragma unroll
    for (int i = 0; i < 4; ++i) {
      int idx = threadIdx.x + i * 256;   // 0..1023 chunk tasks (128 rows x 8)
      int r = idx >> 3, seg = idx & 7;
      copy16_to_lds(&As[buf][r * 72 + seg * 8],
                    A + (size_t)(m0 + r) * K + k0 + seg * 8);
      copy16_to_lds(&Bs[buf][r * 72 + seg * 8],
                    Bt + (size_t)(n0 + r) * K + k0 + seg * 8);
    }
  };

  const v8f vz = {0.f, 0.f, 0.f, 0.f, 0.f, 0.f, 0.f, 0.f};
  v8f acc[4][2];
#pragma unroll
  for (int i = 0; i < 4; ++i)
#pragma unroll
    for (int j = 0; j < 2; ++j) acc[i][j] = vz;

  load_tiles(0, 0);
  int buf = 0;
  for (int k0 = 0; k0 < K; k0 += 64) {
    async_wait_all();
    __syncthreads();
    if (k0 + 64 < K) load_tiles(k0 + 64, buf ^ 1);

#pragma unroll
    for (int ks = 0; ks < 2; ++ks) {
      v16bf af[4];
#pragma unroll
      for (int fm = 0; fm < 4; ++fm)
        af[fm] = load_frag_a(&As[buf][(wm * 64 + fm * 16) * 72 + ks * 32], 72, lane);
#pragma unroll
      for (int fn = 0; fn < 2; ++fn) {
        v16bf bf = load_frag_b(&Bs[buf][(wn * 32 + fn * 16) * 72 + ks * 32], 72, lane);
#pragma unroll
        for (int fm = 0; fm < 4; ++fm) acc[fm][fn] = wmma_bf(af[fm], bf, acc[fm][fn]);
      }
    }
    buf ^= 1;
  }

  // Epilogue. C layout: vgpr r, lane -> row = r + (lane>=16 ? 8:0), col = lane%16.
  const int rbase = m0 + wm * 64 + ((lane & 16) ? 8 : 0);
#pragma unroll
  for (int fn = 0; fn < 2; ++fn) {
    const int col = n0 + wn * 32 + fn * 16 + (lane & 15);
    float bv = 0.f;
    if (HAS_BIAS) bv = bias[col];
#pragma unroll
    for (int fm = 0; fm < 4; ++fm) {
#pragma unroll
      for (int r = 0; r < 8; ++r) {
        const int row = rbase + fm * 16 + r;
        float v = acc[fm][fn][r] + bv;
        if (HAS_RES) v += resid[(size_t)row * ldc + col];
        if (RELU) v = v > 0.f ? v : 0.f;
        if (OUT_BF16) ((bf16_t*)Cout)[(size_t)row * ldc + col] = (bf16_t)v;
        else          ((float*)Cout)[(size_t)row * ldc + col] = v;
      }
    }
  }
}

// ---------------------------------------------------------------------------
// Flash attention. Grid (SEQ/64, NHEAD, B). Block 128 = 4 waves; each wave
// owns 16 query rows. K/V tiles double-buffered in LDS via async copies.
// qkv: [tok][1536] bf16; vtg: [b][h][d][n] bf16 (pre-transposed V).
// ---------------------------------------------------------------------------
__global__ __launch_bounds__(128, 2) void attn_kernel(const bf16_t* __restrict__ qkv,
                                                      const bf16_t* __restrict__ vtg,
                                                      bf16_t* __restrict__ attn_out) {
  __shared__ __align__(16) bf16_t Ks[2][64 * 72];    // [key][d]
  __shared__ __align__(16) bf16_t Vs[2][64 * 72];    // [d][key]
  __shared__ __align__(16) bf16_t Ps[4][16 * 72];    // per-wave P staging

  const int lane = threadIdx.x & 31;
  const int wave = threadIdx.x >> 5;
  const int bb = blockIdx.z, hh = blockIdx.y;
  const int q0 = blockIdx.x * 64;
  const size_t tok0 = (size_t)bb * SEQ;
  const bf16_t* vhead = vtg + (size_t)(bb * NHEAD + hh) * HDIM * SEQ;

  auto load_kv = [&](int kt, int buf) {
#pragma unroll
    for (int i = 0; i < 4; ++i) {
      int idx = threadIdx.x + i * 128;  // 0..511 (64 rows x 8 chunks)
      int r = idx >> 3, seg = idx & 7;
      copy16_to_lds(&Ks[buf][r * 72 + seg * 8],
                    qkv + (tok0 + kt + r) * QKV_LD + CDIM + hh * HDIM + seg * 8);
      copy16_to_lds(&Vs[buf][r * 72 + seg * 8],
                    vhead + (size_t)r * SEQ + kt + seg * 8);
    }
  };

  // Q fragments (d = 64 -> two K=32 steps) straight from global.
  const bf16_t* qbase = qkv + (tok0 + q0 + wave * 16) * QKV_LD + hh * HDIM;
  v16bf qf[2];
  qf[0] = load_frag_a(qbase + 0,  QKV_LD, lane);
  qf[1] = load_frag_a(qbase + 32, QKV_LD, lane);

  const v8f vz = {0.f, 0.f, 0.f, 0.f, 0.f, 0.f, 0.f, 0.f};
  v8f O[4];
  float m_run[8], l_run[8];
#pragma unroll
  for (int i = 0; i < 4; ++i) O[i] = vz;
#pragma unroll
  for (int r = 0; r < 8; ++r) { m_run[r] = -1e30f; l_run[r] = 0.f; }

  load_kv(0, 0);
  int buf = 0;
  for (int kt = 0; kt < SEQ; kt += 64) {
    async_wait_all();
    __syncthreads();
    if (kt + 64 < SEQ) load_kv(kt + 64, buf ^ 1);

    // S = Q K^T * 1/sqrt(64)
    v8f s[4];
#pragma unroll
    for (int nb = 0; nb < 4; ++nb) {
      v8f c = vz;
      c = wmma_bf(qf[0], load_frag_b(&Ks[buf][nb * 16 * 72 + 0],  72, lane), c);
      c = wmma_bf(qf[1], load_frag_b(&Ks[buf][nb * 16 * 72 + 32], 72, lane), c);
#pragma unroll
      for (int r = 0; r < 8; ++r) c[r] *= 0.125f;
      s[nb] = c;
    }

    // Online softmax; rows live in 16-lane halves (xor 8,4,2,1 reductions).
    float mnew[8], sc[8];
#pragma unroll
    for (int r = 0; r < 8; ++r) {
      float mt = fmaxf(fmaxf(s[0][r], s[1][r]), fmaxf(s[2][r], s[3][r]));
#pragma unroll
      for (int m = 8; m; m >>= 1) mt = fmaxf(mt, __shfl_xor(mt, m, 32));
      mnew[r] = fmaxf(m_run[r], mt);
      sc[r] = __expf(m_run[r] - mnew[r]);
      m_run[r] = mnew[r];
    }
#pragma unroll
    for (int r = 0; r < 8; ++r) {
      float rs = 0.f;
#pragma unroll
      for (int nb = 0; nb < 4; ++nb) {
        float p = __expf(s[nb][r] - mnew[r]);
        s[nb][r] = p;
        rs += p;
      }
#pragma unroll
      for (int m = 8; m; m >>= 1) rs += __shfl_xor(rs, m, 32);
      l_run[r] = l_run[r] * sc[r] + rs;
    }

    // Stage P (C layout -> row-major LDS) for A-fragment reads.
    const int half8 = (lane & 16) ? 8 : 0;
#pragma unroll
    for (int nb = 0; nb < 4; ++nb)
#pragma unroll
      for (int r = 0; r < 8; ++r)
        Ps[wave][(r + half8) * 72 + nb * 16 + (lane & 15)] = (bf16_t)s[nb][r];

    // Rescale O, then O += P @ V.
#pragma unroll
    for (int db = 0; db < 4; ++db)
#pragma unroll
      for (int r = 0; r < 8; ++r) O[db][r] *= sc[r];
#pragma unroll
    for (int ks = 0; ks < 2; ++ks) {
      v16bf pa = load_frag_a(&Ps[wave][0] + ks * 32, 72, lane);
#pragma unroll
      for (int db = 0; db < 4; ++db)
        O[db] = wmma_bf(pa, load_frag_b(&Vs[buf][db * 16 * 72 + ks * 32], 72, lane), O[db]);
    }
    buf ^= 1;
  }

  // Normalize and store attn_out[tok][h*64+d] (bf16).
  const int half8 = (lane & 16) ? 8 : 0;
#pragma unroll
  for (int db = 0; db < 4; ++db) {
    const int col = hh * HDIM + db * 16 + (lane & 15);
#pragma unroll
    for (int r = 0; r < 8; ++r) {
      const int row = q0 + wave * 16 + r + half8;
      attn_out[(tok0 + row) * CDIM + col] = (bf16_t)(O[db][r] / l_run[r]);
    }
  }
}

// ---------------------------------------------------------------------------
// Host launcher
// ---------------------------------------------------------------------------
extern "C" void kernel_launch(void* const* d_in, const int* in_sizes, int n_in,
                              void* d_out, int out_size, void* d_ws, size_t ws_size,
                              hipStream_t stream) {
  const float* x     = (const float*)d_in[0];
  const float* ln1_w = (const float*)d_in[1];
  const float* ln1_b = (const float*)d_in[2];
  const float* wq    = (const float*)d_in[3];
  const float* wkv   = (const float*)d_in[4];
  const float* wp    = (const float*)d_in[5];
  const float* bp    = (const float*)d_in[6];
  const float* ln2_w = (const float*)d_in[7];
  const float* ln2_b = (const float*)d_in[8];
  const float* w1    = (const float*)d_in[9];
  const float* b1    = (const float*)d_in[10];
  const float* w2    = (const float*)d_in[11];
  const float* b2    = (const float*)d_in[12];
  float* out = (float*)d_out;

  char* ws = (char*)d_ws;
  size_t off = 0;
  auto take = [&](size_t bytes) -> char* {
    char* p = ws + off;
    off += (bytes + 255) & ~(size_t)255;
    return p;
  };
  bf16_t* wq_t   = (bf16_t*)take((size_t)CDIM * CDIM * 2);        // [N][K]
  bf16_t* wkv_t  = (bf16_t*)take((size_t)2 * CDIM * CDIM * 2);
  bf16_t* wp_t   = (bf16_t*)take((size_t)CDIM * CDIM * 2);
  bf16_t* w1_t   = (bf16_t*)take((size_t)HIDDEN * CDIM * 2);
  bf16_t* w2_t   = (bf16_t*)take((size_t)CDIM * HIDDEN * 2);
  bf16_t* xn1    = (bf16_t*)take((size_t)TOK * CDIM * 2);
  bf16_t* qkv    = (bf16_t*)take((size_t)TOK * QKV_LD * 2);
  bf16_t* vt     = (bf16_t*)take((size_t)TOK * CDIM * 2);         // [b][h][d][n]
  bf16_t* attn   = (bf16_t*)take((size_t)TOK * CDIM * 2);
  float*  x1     = (float*) take((size_t)TOK * CDIM * 4);
  bf16_t* xn2    = (bf16_t*)take((size_t)TOK * CDIM * 2);
  bf16_t* hbuf   = (bf16_t*)take((size_t)TOK * HIDDEN * 2);

  // 0) weights -> bf16, transposed to [N][K]
  auto cvt = [&](const float* src, bf16_t* dst, int K, int N) {
    cvt_transpose_kernel<<<(K * N + 255) / 256, 256, 0, stream>>>(src, dst, K, N);
  };
  cvt(wq,  wq_t,  CDIM, CDIM);
  cvt(wkv, wkv_t, CDIM, 2 * CDIM);
  cvt(wp,  wp_t,  CDIM, CDIM);
  cvt(w1,  w1_t,  CDIM, HIDDEN);
  cvt(w2,  w2_t,  HIDDEN, CDIM);

  // 1) LN1
  ln_kernel<<<TOK, 256, 0, stream>>>(x, ln1_w, ln1_b, xn1);

  // 2) QKV projections into fused [tok][1536]
  gemm_bf16_kernel<true, false, false, false>
      <<<dim3(CDIM / 128, TOK / 128), 256, 0, stream>>>(
          xn1, wq_t, qkv, nullptr, nullptr, TOK, CDIM, CDIM, QKV_LD);
  gemm_bf16_kernel<true, false, false, false>
      <<<dim3(2 * CDIM / 128, TOK / 128), 256, 0, stream>>>(
          xn1, wkv_t, qkv + CDIM, nullptr, nullptr, TOK, 2 * CDIM, CDIM, QKV_LD);

  // 3) V transpose, then flash attention
  v_transpose_kernel<<<(TOK * CDIM) / 256, 256, 0, stream>>>(qkv, vt);
  attn_kernel<<<dim3(SEQ / 64, NHEAD, 2), 128, 0, stream>>>(qkv, vt, attn);

  // 4) output projection + residual (f32)
  gemm_bf16_kernel<false, false, true, true>
      <<<dim3(CDIM / 128, TOK / 128), 256, 0, stream>>>(
          attn, wp_t, x1, bp, x, TOK, CDIM, CDIM, CDIM);

  // 5) LN2
  ln_kernel<<<TOK, 256, 0, stream>>>(x1, ln2_w, ln2_b, xn2);

  // 6) MLP up + ReLU (bf16)
  gemm_bf16_kernel<true, true, true, false>
      <<<dim3(HIDDEN / 128, TOK / 128), 256, 0, stream>>>(
          xn2, w1_t, hbuf, b1, nullptr, TOK, HIDDEN, CDIM, HIDDEN);

  // 7) MLP down + residual -> d_out (f32)
  gemm_bf16_kernel<false, false, true, true>
      <<<dim3(CDIM / 128, TOK / 128), 256, 0, stream>>>(
          hbuf, w2_t, out, b2, x1, TOK, CDIM, HIDDEN, CDIM);
}
